// MoEFFN_77214922047963
// MI455X (gfx1250) — matrix operand order repaired and checked by reading, not compiled, not run on hardware
//
#include <hip/hip_runtime.h>
#include <hip/hip_bf16.h>
#include <math.h>

#define BT    4096   // B*T tokens
#define DDIM  1024
#define EEXP  8
#define FFDIM 2048

typedef __attribute__((ext_vector_type(16))) __bf16        bf16x16;
typedef __attribute__((ext_vector_type(8)))  float         floatx8;
typedef __attribute__((ext_vector_type(8)))  unsigned int  uintx8;

static __device__ __forceinline__ unsigned short f32_to_bf16(float f) {
    unsigned int u = __builtin_bit_cast(unsigned int, f);
    unsigned int r = u + 0x7FFFu + ((u >> 16) & 1u);   // round-to-nearest-even
    return (unsigned short)(r >> 16);
}

// Async global->LDS copy (CDNA5 ASYNCcnt path); builtins don't cover these.
static __device__ __forceinline__ void async_load_b128(unsigned lds_off, const void* gptr) {
    asm volatile("global_load_async_to_lds_b128 %0, %1, off"
                 :: "v"(lds_off), "v"(gptr) : "memory");
}
static __device__ __forceinline__ void wait_async_le2() {
    asm volatile("s_wait_asynccnt 0x2" ::: "memory");
}
static __device__ __forceinline__ void wait_async_0() {
    asm volatile("s_wait_asynccnt 0x0" ::: "memory");
}

// ---------------------------------------------------------------- init ----
__global__ void init_kernel(float* __restrict__ out, long n, int* __restrict__ head) {
    long i = (long)blockIdx.x * blockDim.x + threadIdx.x;
    long stride = (long)gridDim.x * blockDim.x;
    for (; i < n; i += stride) out[i] = 0.0f;
    if (blockIdx.x == 0 && threadIdx.x < 64) head[threadIdx.x] = 0;
}

// ------------------------------------------------------------- convert ----
__global__ void cvt_bf16_kernel(const float* __restrict__ src,
                                unsigned short* __restrict__ dst, int n) {
    int i = blockIdx.x * blockDim.x + threadIdx.x;
    int stride = gridDim.x * blockDim.x;
    for (; i < n; i += stride) dst[i] = f32_to_bf16(src[i]);
}

// Pack weights into WMMA pair-interleaved bf16 layout:
// dst[e][p][n] (u32) = { bf16(W[e][2p][n]) , bf16(W[e][2p+1][n]) }
__global__ void cvt_pack_kernel(const float* __restrict__ src,
                                unsigned int* __restrict__ dst,
                                int halfK, int Nd, int total) {
    int i = blockIdx.x * blockDim.x + threadIdx.x;
    int stride = gridDim.x * blockDim.x;
    for (; i < total; i += stride) {
        int n = i % Nd;
        int p = (i / Nd) % halfK;
        int e = i / (Nd * halfK);
        long base = ((long)e * (2 * halfK) + 2 * p) * Nd + n;
        unsigned int lo = f32_to_bf16(src[base]);
        unsigned int hi = f32_to_bf16(src[base + Nd]);
        dst[i] = lo | (hi << 16);
    }
}

// -------------------------------------------------------------- router ----
__global__ __launch_bounds__(256)
void router_kernel(const float* __restrict__ x, const float* __restrict__ gate_w,
                   int* __restrict__ assign_cnt, int* __restrict__ top1_cnt,
                   float* __restrict__ probs_sum, float* __restrict__ sumsq,
                   int* __restrict__ token_list, float* __restrict__ gate_wt) {
    __shared__ float sg[EEXP * DDIM];                       // 32 KB LDS-resident gate
    for (int i = threadIdx.x; i < EEXP * DDIM / 4; i += blockDim.x)
        ((float4*)sg)[i] = ((const float4*)gate_w)[i];
    __syncthreads();

    const int wv   = threadIdx.x >> 5;
    const int lane = threadIdx.x & 31;
    const int t    = blockIdx.x * 8 + wv;                   // one wave32 per token

    float acc[EEXP];
#pragma unroll
    for (int e = 0; e < EEXP; ++e) acc[e] = 0.0f;
    const float* xr = x + (long)t * DDIM;
    for (int d = lane; d < DDIM; d += 32) {
        float xv = xr[d];
#pragma unroll
        for (int e = 0; e < EEXP; ++e) acc[e] += xv * sg[e * DDIM + d];
    }
#pragma unroll
    for (int off = 16; off > 0; off >>= 1)
#pragma unroll
        for (int e = 0; e < EEXP; ++e) acc[e] += __shfl_down(acc[e], off, 32);

    if (lane == 0) {
        int i0 = 0; float l0 = acc[0];
        for (int e = 1; e < EEXP; ++e) if (acc[e] > l0) { l0 = acc[e]; i0 = e; }
        int i1 = (i0 == 0) ? 1 : 0; float l1 = acc[i1];
        for (int e = 0; e < EEXP; ++e)
            if (e != i0 && acc[e] > l1) { l1 = acc[e]; i1 = e; }
        float mx = fmaxf(l0, l1);
        float e0 = expf(l0 - mx), e1 = expf(l1 - mx);
        float inv = 1.0f / (e0 + e1);
        float w0 = e0 * inv, w1 = e1 * inv;

        float pm = acc[0];
        for (int e = 1; e < EEXP; ++e) pm = fmaxf(pm, acc[e]);
        float ps = 0.0f, sq = 0.0f, pe[EEXP];
        for (int e = 0; e < EEXP; ++e) {
            pe[e] = expf(acc[e] - pm); ps += pe[e]; sq += acc[e] * acc[e];
        }
        float pinv = 1.0f / ps;
        for (int e = 0; e < EEXP; ++e) atomicAdd(&probs_sum[e], pe[e] * pinv);
        atomicAdd(sumsq, sq);
        atomicAdd(&top1_cnt[i0], 1);

        int p0 = atomicAdd(&assign_cnt[i0], 1);
        token_list[i0 * BT + p0] = t; gate_wt[i0 * BT + p0] = w0;
        int p1 = atomicAdd(&assign_cnt[i1], 1);
        token_list[i1 * BT + p1] = t; gate_wt[i1 * BT + p1] = w1;
    }
}

// ---------------------------------------------------------------- scan ----
__global__ void scan_kernel(const int* __restrict__ assign_cnt,
                            int* __restrict__ offsets, int* __restrict__ rounded) {
    if (threadIdx.x == 0 && blockIdx.x == 0) {
        int off = 0;
        for (int e = 0; e < EEXP; ++e) {
            offsets[e] = off;
            int rc = (assign_cnt[e] + 15) & ~15;
            rounded[e] = rc;
            off += rc;
        }
    }
}

// --------------------------------------------------------- fused GEMMs ----
// Block: 256 thr = 8 waves as 4(M) x 2(N). Tile M=64, N=128.
// Each wave: 16M x 64N = 4 accumulators -> 4 v_wmma per K-step.
// B tile (32K x 128N, pair-packed u32) double-buffered in LDS via
// global_load_async_to_lds_b128 (ASYNCcnt), overlapping copy with compute.
template <bool FFN1>
__global__ __launch_bounds__(256)
void moe_gemm_kernel(const unsigned short* __restrict__ Abase,   // x_bf16 | H (row-major bf16)
                     const unsigned int*  __restrict__ Bpack,    // pair-packed w1 | w2
                     const float* __restrict__ bias,             // b1 | b2  [E, N]
                     const int* __restrict__ counts,
                     const int* __restrict__ offsets,
                     const int* __restrict__ rounded,
                     const int* __restrict__ token_list,
                     const float* __restrict__ gate_wt,
                     unsigned short* __restrict__ H,
                     float* __restrict__ out,
                     int Kdim, int Ndim) {
    const int e  = blockIdx.z;
    const int m0 = blockIdx.y * 64;
    const int rc = rounded[e];
    if (m0 >= rc) return;
    const int cnt = counts[e];
    const int n0  = blockIdx.x * 128;

    const int tid  = threadIdx.x;
    const int lane = tid & 31;
    const int wv   = tid >> 5;
    const int wm   = wv & 3;              // M-subtile 0..3
    const int wn   = wv >> 2;             // N-half    0..1
    const int col  = lane & 15;
    const int half = lane >> 4;

    __shared__ unsigned int sB[2][16 * 128];   // 2 x 8KB pair-packed B tiles
    __shared__ int   s_tok[64];
    __shared__ float s_wt[64];

    if (tid < 64) {
        int idx = m0 + tid;
        int tok = 0; float w = 0.0f;
        if (idx < cnt) { tok = token_list[e * BT + idx]; w = gate_wt[e * BT + idx]; }
        s_tok[tid] = tok; s_wt[tid] = w;
    }
    __syncthreads();

    // A-fragment row for this lane (16-bit A layout: rows 0..15 on both halves)
    long arow;
    if (FFN1) {
        arow = (long)s_tok[wm * 16 + col];
    } else {
        int lr = m0 + wm * 16 + col;
        if (lr > rc - 1) lr = rc - 1;              // clamp inside H region
        arow = (long)offsets[e] + lr;
    }
    const unsigned short* Arow = Abase + arow * (long)Kdim;
    const unsigned int*   Bexp = Bpack + (long)e * (Kdim >> 1) * Ndim;

    const unsigned sB_base = (unsigned)(unsigned long long)&sB[0][0]; // LDS byte offset

    // stage one 32K x 128N pair-packed tile: 16 rows x 512B, 512 b128 chunks
    auto stage = [&](int k0, int buf) {
        const unsigned int* gsrc = Bexp + (long)(k0 >> 1) * Ndim + n0;
        unsigned lbase = sB_base + (unsigned)buf * (16 * 128 * 4);
#pragma unroll
        for (int j = 0; j < 2; ++j) {
            int c    = tid + j * 256;          // 0..511
            int row  = c >> 5;                 // 0..15
            int boff = (c & 31) * 16;          // byte offset in 512B row
            const char* g = (const char*)(gsrc + (long)row * Ndim) + boff;
            async_load_b128(lbase + (unsigned)(row * 512 + boff), g);
        }
    };

    floatx8 acc[4];
#pragma unroll
    for (int s = 0; s < 4; ++s) acc[s] = (floatx8){0.f,0.f,0.f,0.f,0.f,0.f,0.f,0.f};

    const int nk = Kdim >> 5;                  // K-steps of 32
    stage(0, 0);
    for (int ks = 0; ks < nk; ++ks) {
        const int k0 = ks << 5;
        if (ks + 1 < nk) {
            stage(k0 + 32, (ks + 1) & 1);      // prefetch next tile into other buffer
            wait_async_le2();                  // in-order: current tile has landed
        } else {
            wait_async_0();
        }
        __syncthreads();                       // tile ks visible to all waves

        // A fragment: V0-3 = K(half*8 + 0..7), V4-7 = K(16 + half*8 + 0..7)
        uintx8 au;
        {
            const uint4 lo = *(const uint4*)(Arow + k0 + half * 8);
            const uint4 hi = *(const uint4*)(Arow + k0 + 16 + half * 8);
            au[0] = lo.x; au[1] = lo.y; au[2] = lo.z; au[3] = lo.w;
            au[4] = hi.x; au[5] = hi.y; au[6] = hi.z; au[7] = hi.w;
        }
        bf16x16 av = __builtin_bit_cast(bf16x16, au);

        const unsigned int* sb = &sB[ks & 1][0];
#pragma unroll
        for (int s = 0; s < 4; ++s) {
            // B fragment: Vv = K-pair (v + half*4 + (v & 4)), column wn*64 + s*16 + col
            uintx8 bu;
#pragma unroll
            for (int v = 0; v < 8; ++v)
                bu[v] = sb[(v + half * 4 + (v & 4)) * 128 + wn * 64 + s * 16 + col];
            bf16x16 bv = __builtin_bit_cast(bf16x16, bu);
            acc[s] = __builtin_amdgcn_wmma_f32_16x16x32_bf16(false, av, false, bv,
                                                             (short)0, acc[s], false, false);
        }
        __syncthreads();                       // done reading before buffer reuse
    }

    // ---- epilogue: C/D layout -> VGPR r holds M = r + half*8, N = col ----
#pragma unroll
    for (int s = 0; s < 4; ++s) {
        const int n = n0 + wn * 64 + s * 16 + col;
        const float b = bias[e * Ndim + n];
        if (FFN1) {
            long hrow0 = (long)offsets[e] + m0;
#pragma unroll
            for (int r = 0; r < 8; ++r) {
                int lr = wm * 16 + r + half * 8;
                if (m0 + lr < rc) {
                    float h = acc[s][r] + b;
                    float g = 0.5f * h * (1.0f + erff(h * 0.70710678118654752f));
                    H[(hrow0 + lr) * (long)FFDIM + n] = f32_to_bf16(g);
                }
            }
        } else {
#pragma unroll
            for (int r = 0; r < 8; ++r) {
                int lr = wm * 16 + r + half * 8;
                if (m0 + lr < cnt) {
                    float y = acc[s][r] + b;
                    atomicAdd(&out[(long)s_tok[lr] * DDIM + n], s_wt[lr] * y);
                }
            }
        }
    }
}

// ----------------------------------------------------------------- aux ----
__global__ void aux_kernel(const int* __restrict__ top1_cnt,
                           const float* __restrict__ probs_sum,
                           const float* __restrict__ sumsq,
                           float* __restrict__ out_aux) {
    if (threadIdx.x == 0 && blockIdx.x == 0) {
        float s = 0.0f;
        for (int e = 0; e < EEXP; ++e)
            s += ((float)top1_cnt[e] / (float)BT) * (probs_sum[e] / (float)BT);
        *out_aux = (float)EEXP * s + (sumsq[0] / ((float)BT * EEXP)) * 0.001f;
    }
}

// -------------------------------------------------------------- launch ----
extern "C" void kernel_launch(void* const* d_in, const int* in_sizes, int n_in,
                              void* d_out, int out_size, void* d_ws, size_t ws_size,
                              hipStream_t stream) {
    const float* x      = (const float*)d_in[0];
    const float* gate_w = (const float*)d_in[1];
    const float* w1     = (const float*)d_in[2];
    const float* w2     = (const float*)d_in[3];
    const float* b1     = (const float*)d_in[4];
    const float* b2     = (const float*)d_in[5];
    float* out = (float*)d_out;

    char* ws = (char*)d_ws;
    int*   head       = (int*)ws;            // 64 ints (zeroed each call)
    int*   assign_cnt = head + 0;
    int*   top1_cnt   = head + 8;
    int*   offsets    = head + 16;
    int*   rounded    = head + 24;
    float* probs_sum  = (float*)(head + 32);
    float* sumsq      = (float*)(head + 40);

    size_t off = 256;
    int*   token_list = (int*)(ws + off);              off += (size_t)EEXP * BT * 4;
    float* gate_wt    = (float*)(ws + off);            off += (size_t)EEXP * BT * 4;
    unsigned short* xb  = (unsigned short*)(ws + off); off += (size_t)BT * DDIM * 2;
    unsigned int* w1p = (unsigned int*)(ws + off);     off += (size_t)EEXP * (DDIM / 2) * FFDIM * 4;
    unsigned int* w2p = (unsigned int*)(ws + off);     off += (size_t)EEXP * (FFDIM / 2) * DDIM * 4;
    unsigned short* Hbuf = (unsigned short*)(ws + off); // (2*BT + 16*E) rows x FF bf16

    const long n_out = (long)BT * DDIM + 1;
    init_kernel<<<1024, 256, 0, stream>>>(out, n_out, head);
    cvt_bf16_kernel<<<2048, 256, 0, stream>>>(x, xb, BT * DDIM);
    cvt_pack_kernel<<<4096, 256, 0, stream>>>(w1, w1p, DDIM / 2, FFDIM,
                                              EEXP * (DDIM / 2) * FFDIM);
    cvt_pack_kernel<<<4096, 256, 0, stream>>>(w2, w2p, FFDIM / 2, DDIM,
                                              EEXP * (FFDIM / 2) * DDIM);

    router_kernel<<<BT / 8, 256, 0, stream>>>(x, gate_w, assign_cnt, top1_cnt,
                                              probs_sum, sumsq, token_list, gate_wt);
    scan_kernel<<<1, 1, 0, stream>>>(assign_cnt, offsets, rounded);

    dim3 g1(FFDIM / 128, BT / 64, EEXP);
    moe_gemm_kernel<true><<<g1, 256, 0, stream>>>(xb, w1p, b1, assign_cnt, offsets,
                                                  rounded, token_list, gate_wt,
                                                  Hbuf, out, DDIM, FFDIM);
    dim3 g2(DDIM / 128, BT / 64, EEXP);
    moe_gemm_kernel<false><<<g2, 256, 0, stream>>>(Hbuf, w2p, b2, assign_cnt, offsets,
                                                   rounded, token_list, gate_wt,
                                                   Hbuf, out, FFDIM, DDIM);

    aux_kernel<<<1, 1, 0, stream>>>(top1_cnt, probs_sum, sumsq, out + (long)BT * DDIM);
}
